// Model_9191230013710
// MI455X (gfx1250) — matrix operand order repaired and checked by reading, not compiled, not run on hardware
//
#include <hip/hip_runtime.h>
#include <hip/hip_bf16.h>

// ---------------------------------------------------------------------------
// MI455X (gfx1250) implementation.
// Compute-dominant parts (BiLSTM input GEMMs, LSTM recurrences, MLP head) run
// on V_WMMA_F32_16X16X32_F16 (f16 in / f32 accumulate). Graph + GRU stages are
// VALU/atomic (tiny matvecs, scatter-adds -> no matrix-core benefit).
// ---------------------------------------------------------------------------

typedef __attribute__((ext_vector_type(16))) _Float16 v16h;
typedef __attribute__((ext_vector_type(8)))  _Float16 v8h;
typedef __attribute__((ext_vector_type(8)))  float    v8f;

__device__ __forceinline__ float sigm(float x) { return 1.0f / (1.0f + __expf(-x)); }

__device__ __forceinline__ v8f wmma_f16(v16h a, v16h b, v8f c) {
    // D = A(16x32 f16) * B(32x16 f16) + C(16x16 f32)
    return __builtin_amdgcn_wmma_f32_16x16x32_f16(false, a, false, b, (short)0, c, false, false);
}

// ---------------------------------------------------------------------------
// Model dims
// ---------------------------------------------------------------------------
#define HID   20
#define LH    100
#define HP    112          // LH padded to 7*16
#define NP    448          // 4 gates * HP
#define SEQ   144
#define NN    20000
#define NE    640000
#define BSZ   1024
#define MTOT  (BSZ * SEQ)  // 147456

#define WHH_HALVES (4 * 28 * 32 * 16)   // 57344 halves = 112KB packed Whh
#define HL_OFF     WHH_HALVES
#define HL_SIZE    (4 * 16 * 128)       // 8192 halves = 16KB transpose tiles
#define RECUR_SMEM ((WHH_HALVES + HL_SIZE) * 2)  // 131072 bytes

// ---------------------------------------------------------------------------
// Small utility kernels
// ---------------------------------------------------------------------------
__global__ void zero_f32(float* p, long n) {
    long i = (long)blockIdx.x * blockDim.x + threadIdx.x;
    if (i < n) p[i] = 0.0f;
}

// ---------------------------------------------------------------------------
// Stage 1: bidirectional GRU encoder over node time series (per-thread, f32).
// Writes hf flat [NN*HID] then hb flat [NN*HID]  (== raw reshape of [2,N,H]).
// ---------------------------------------------------------------------------
__global__ void gru_kernel(const float* __restrict__ gx_in,
                           const float* __restrict__ WhhF, const float* __restrict__ WihF,
                           const float* __restrict__ bhhF, const float* __restrict__ bihF,
                           const float* __restrict__ WhhB, const float* __restrict__ WihB,
                           const float* __restrict__ bhhB, const float* __restrict__ bihB,
                           float* __restrict__ hnb) {
    __shared__ float sWhh[2][3 * HID * HID];   // [60][20]
    __shared__ float sWih[2][3 * HID * 2];     // [60][2]
    __shared__ float sbh[2][3 * HID];
    __shared__ float sbi[2][3 * HID];
    for (int i = threadIdx.x; i < 3 * HID * HID; i += blockDim.x) { sWhh[0][i] = WhhF[i]; sWhh[1][i] = WhhB[i]; }
    for (int i = threadIdx.x; i < 3 * HID * 2;   i += blockDim.x) { sWih[0][i] = WihF[i]; sWih[1][i] = WihB[i]; }
    for (int i = threadIdx.x; i < 3 * HID;       i += blockDim.x) {
        sbh[0][i] = bhhF[i]; sbh[1][i] = bhhB[i];
        sbi[0][i] = bihF[i]; sbi[1][i] = bihB[i];
    }
    __syncthreads();
    int n = blockIdx.x * blockDim.x + threadIdx.x;
    if (n >= NN) return;

    for (int d = 0; d < 2; ++d) {
        float h[HID];
#pragma unroll
        for (int j = 0; j < HID; ++j) h[j] = 0.0f;
        for (int s = 0; s < SEQ; ++s) {
            int t = d ? (SEQ - 1 - s) : s;
            float x0 = gx_in[((long)n * SEQ + t) * 2 + 0];
            float x1 = gx_in[((long)n * SEQ + t) * 2 + 1];
            float hn2[HID];
#pragma unroll
            for (int j = 0; j < HID; ++j) {
                float gr = sbi[d][j]            + sWih[d][j * 2] * x0            + sWih[d][j * 2 + 1] * x1;
                float gz = sbi[d][HID + j]      + sWih[d][(HID + j) * 2] * x0     + sWih[d][(HID + j) * 2 + 1] * x1;
                float gn = sbi[d][2 * HID + j]  + sWih[d][(2 * HID + j) * 2] * x0 + sWih[d][(2 * HID + j) * 2 + 1] * x1;
                float hr = sbh[d][j], hz = sbh[d][HID + j], hnv = sbh[d][2 * HID + j];
#pragma unroll
                for (int k = 0; k < HID; ++k) {
                    hr  += sWhh[d][j * HID + k] * h[k];
                    hz  += sWhh[d][(HID + j) * HID + k] * h[k];
                    hnv += sWhh[d][(2 * HID + j) * HID + k] * h[k];
                }
                float r = sigm(gr + hr);
                float z = sigm(gz + hz);
                float nn = tanhf(gn + r * hnv);
                hn2[j] = (1.0f - z) * nn + z * h[j];
            }
#pragma unroll
            for (int j = 0; j < HID; ++j) h[j] = hn2[j];
        }
        float* out = hnb + (long)d * NN * HID + (long)n * HID;
#pragma unroll
        for (int j = 0; j < HID; ++j) out[j] = h[j];
    }
}

// x = relu(hn[N,40] @ W[40,20] + b)
__global__ void lstm_fc_kernel(const float* __restrict__ hn, const float* __restrict__ W,
                               const float* __restrict__ b, float* __restrict__ out) {
    int n = blockIdx.x * blockDim.x + threadIdx.x;
    if (n >= NN) return;
    for (int j = 0; j < HID; ++j) {
        float acc = b[j];
        for (int k = 0; k < 2 * HID; ++k) acc += hn[(long)n * 2 * HID + k] * W[k * HID + j];
        out[(long)n * HID + j] = fmaxf(acc, 0.0f);
    }
}

// ---------------------------------------------------------------------------
// ARMA graph stage
// ---------------------------------------------------------------------------
__global__ void deg_kernel(const int* __restrict__ ei, const float* __restrict__ ew,
                           float* __restrict__ deg) {
    int e = blockIdx.x * blockDim.x + threadIdx.x;
    if (e >= NE) return;
    atomicAdd(&deg[ei[NE + e]], ew[e]);
}
__global__ void dinv_kernel(float* __restrict__ deg) {
    int n = blockIdx.x * blockDim.x + threadIdx.x;
    if (n >= NN) return;
    float d = deg[n];
    deg[n] = (d > 0.0f) ? rsqrtf(d) : 0.0f;
}
__global__ void norm_kernel(const int* __restrict__ ei, const float* __restrict__ ew,
                            const float* __restrict__ dinv, float* __restrict__ nrm) {
    int e = blockIdx.x * blockDim.x + threadIdx.x;
    if (e >= NE) return;
    nrm[e] = dinv[ei[e]] * ew[e] * dinv[ei[NE + e]];
}
// h = x @ W  ([NN,20] @ [20,20], W row-major K x N)
__global__ void matvec20_kernel(const float* __restrict__ x, const float* __restrict__ W,
                                float* __restrict__ h) {
    int n = blockIdx.x * blockDim.x + threadIdx.x;
    if (n >= NN) return;
    for (int j = 0; j < HID; ++j) {
        float acc = 0.0f;
        for (int k = 0; k < HID; ++k) acc += x[(long)n * HID + k] * W[k * HID + j];
        h[(long)n * HID + j] = acc;
    }
}
__global__ void scatter_kernel(const int* __restrict__ ei, const float* __restrict__ nrm,
                               const float* __restrict__ h, float* __restrict__ agg) {
    int e = blockIdx.x * blockDim.x + threadIdx.x;
    if (e >= NE) return;
    int r = ei[e], c = ei[NE + e];
    float nv = nrm[e];
    for (int j = 0; j < HID; ++j) atomicAdd(&agg[(long)c * HID + j], nv * h[(long)r * HID + j]);
}
// out = relu(agg + x @ V + b)
__global__ void arma_combine_kernel(const float* __restrict__ x, const float* __restrict__ agg,
                                    const float* __restrict__ V, const float* __restrict__ b,
                                    float* __restrict__ out) {
    int n = blockIdx.x * blockDim.x + threadIdx.x;
    if (n >= NN) return;
    for (int j = 0; j < HID; ++j) {
        float acc = b[j] + agg[(long)n * HID + j];
        for (int k = 0; k < HID; ++k) acc += x[(long)n * HID + k] * V[k * HID + j];
        out[(long)n * HID + j] = fmaxf(acc, 0.0f);
    }
}

// ---------------------------------------------------------------------------
// Build BiLSTM layer-1 input: f16 [B*SEQ, 64] = [fc(data_feat) | emb[ids] | 0]
// ---------------------------------------------------------------------------
__global__ void build_in_kernel(const float* __restrict__ df, const int* __restrict__ ids,
                                const float* __restrict__ emb, const float* __restrict__ fcW,
                                const float* __restrict__ fcb, _Float16* __restrict__ buf0) {
    int idx = blockIdx.x * blockDim.x + threadIdx.x;
    if (idx >= MTOT) return;
    int b = idx / SEQ;
    float x0 = df[(long)idx * 2], x1 = df[(long)idx * 2 + 1];
    _Float16* o = buf0 + (long)idx * 64;
    for (int j = 0; j < HID; ++j)
        o[j] = (_Float16)(fcb[j] + x0 * fcW[j] + x1 * fcW[HID + j]);
    int node = ids[b];
    for (int j = 0; j < HID; ++j) o[HID + j] = (_Float16)emb[(long)node * HID + j];
    for (int j = 2 * HID; j < 64; ++j) o[j] = (_Float16)0.0f;
}

// ---------------------------------------------------------------------------
// Weight packing into fragment-contiguous f16 B layout.
// B fragment for (kt,nt): lane holds col n = nt*16 + (lane&15),
//   K = kt*32 + (lane>>4)*16 + j  for 16 contiguous halves j=0..15.
// Gate remap (G>0): padded col np -> gate g=np/Hp, j=np%Hp (valid j<H).
// K remap: ksrc = kp<kA ? kp : (kSegB<=kp<kSegB+kB ? kA+(kp-kSegB) : invalid).
// orient==0: W stored [Nrows, Kreal] (Wih/Whh: B = W^T). orient==1: W stored [K, N].
// ---------------------------------------------------------------------------
__global__ void pack_b_kernel(const float* __restrict__ W, _Float16* __restrict__ out,
                              int KT, int NT, int G, int Hp, int H, int Nreal,
                              int kA, int kSegB, int kB, int ld, int orient) {
    int idx = blockIdx.x * blockDim.x + threadIdx.x;
    int total = KT * NT * 512;
    if (idx >= total) return;
    int j = idx & 15;
    int lane = (idx >> 4) & 31;
    int tile = idx >> 9;
    int nt = tile % NT, kt = tile / NT;
    int np = nt * 16 + (lane & 15);
    int kp = kt * 32 + ((lane >> 4) << 4) + j;
    int nsrc; bool nok;
    if (G > 0) { int g = np / Hp; int jj = np % Hp; nok = (jj < H); nsrc = g * H + jj; }
    else       { nok = (np < Nreal); nsrc = np; }
    int ksrc = -1;
    if (kp < kA) ksrc = kp;
    else if (kB > 0 && kp >= kSegB && kp < kSegB + kB) ksrc = kA + (kp - kSegB);
    float v = 0.0f;
    if (nok && ksrc >= 0)
        v = orient ? W[(long)ksrc * ld + nsrc] : W[(long)nsrc * ld + ksrc];
    out[idx] = (_Float16)v;
}

__global__ void pack_bias_kernel(const float* __restrict__ b, float* __restrict__ out,
                                 int Np, int G, int Hp, int H, int Nreal) {
    int i = blockIdx.x * blockDim.x + threadIdx.x;
    if (i >= Np) return;
    float v = 0.0f;
    if (G > 0) { int g = i / Hp, jj = i % Hp; if (jj < H) v = b[g * H + jj]; }
    else if (i < Nreal) v = b[i];
    out[i] = v;
}

// ---------------------------------------------------------------------------
// Generic WMMA GEMM: each wave computes a 16 x (NTILES*16) output tile.
// Templated on KT (K-chunks of 32) and NTILES (1 or 4 active n-tiles) so the
// hot loop is branch-free: NTILES back-to-back v_wmma per K-chunk, in-place
// accumulation (no exec-mask merges / register shuffles).
// A: f16 row-major [Mtot, lda] (lda multiple of 32, zero-padded K).
// Bp: packed f16 fragments.  C = A*B + bias, optional ReLU.
// Output: f32 [rows, ldc] (only n < nvalid; optional (b,t)->(t,b) row transpose)
//      or f16 [Mtot, ldc] (cols >= nvalid stored as 0).
// ---------------------------------------------------------------------------
template <int KT, int NTILES>
__global__ void gemm16_kernel(const _Float16* __restrict__ A, int lda,
                              const _Float16* __restrict__ Bp,
                              const float* __restrict__ biasP,
                              float* __restrict__ outF32, _Float16* __restrict__ outF16,
                              int Mtot, int NT, int nvalid, int ldc,
                              int relu, int transp, int Tdim, int Bdim) {
    int n4cnt = (NT + 3) >> 2;
    int gwave = blockIdx.x * (blockDim.x >> 5) + (threadIdx.x >> 5);
    int MT = Mtot >> 4;
    if (gwave >= MT * n4cnt) return;
    int mt = gwave / n4cnt, n4 = gwave % n4cnt;
    int lane = threadIdx.x & 31;
    int rowA = mt * 16 + (lane & 15);
    int kbase = (lane >> 4) * 16;

    v8f acc[NTILES];
#pragma unroll
    for (int i = 0; i < NTILES; ++i)
#pragma unroll
        for (int r = 0; r < 8; ++r) acc[i][r] = 0.0f;

    const _Float16* arow = A + (long)rowA * lda + kbase;
    const _Float16* bbase = Bp + ((long)n4 * 4 * 32) * 16 + (long)lane * 16;

#pragma unroll
    for (int kc = 0; kc < KT; ++kc) {
        const _Float16* ap = arow + kc * 32;
        v8h lo = *(const v8h*)ap;
        v8h hi = *(const v8h*)(ap + 8);
        v16h a;
#pragma unroll
        for (int j = 0; j < 8; ++j) { a[j] = lo[j]; a[8 + j] = hi[j]; }
#pragma unroll
        for (int i = 0; i < NTILES; ++i) {
            v16h bf = *(const v16h*)(bbase + ((long)kc * NT + i) * 512);
            acc[i] = wmma_f16(a, bf, acc[i]);
        }
    }
    int mbase = mt * 16 + (lane >> 4) * 8;
#pragma unroll
    for (int i = 0; i < NTILES; ++i) {
        int n = (n4 * 4 + i) * 16 + (lane & 15);
        float bv = biasP ? biasP[n] : 0.0f;
#pragma unroll
        for (int r = 0; r < 8; ++r) {
            float v = acc[i][r] + bv;
            if (relu) v = fmaxf(v, 0.0f);
            int m = mbase + r;
            if (outF16) {
                outF16[(long)m * ldc + n] = (_Float16)(n < nvalid ? v : 0.0f);
            } else if (n < nvalid) {
                long row = transp ? (long)(m % Tdim) * Bdim + (m / Tdim) : (long)m;
                outF32[row * ldc + n] = v;
            }
        }
    }
}

// ---------------------------------------------------------------------------
// LSTM recurrence. 64 waves, each owns 16 batch rows for the whole sequence.
// Packed Whh (112KB) is staged once into LDS (dynamic shared, 320KB/WGP on
// CDNA5) and re-read every step via ds_load; an asm-laundered offset stops
// LLVM from hoisting the loop-invariant fragment loads into (spilled) regs.
// h kept as four 16x32 f16 A-fragments; c as seven f32 D-tiles (registers).
// Per step: 7 jt-tiles x 4 gates x 4 k-chunks = 112 v_wmma per wave.
// h_new D->A transpose goes through a wave-private zero-padded LDS tile.
// gx: f32 [SEQ][BSZ][NP]. out: f16 [BSZ][SEQ][224] (+colOff), padded cols = 0.
// ---------------------------------------------------------------------------
__global__ void __launch_bounds__(128)
lstm_recur_kernel(const float* __restrict__ gx,
                  const _Float16* __restrict__ WhhP,
                  _Float16* __restrict__ outbuf,
                  int colOff, int reverse) {
    extern __shared__ _Float16 smem[];   // [0,57344): Whh frags; [57344,+8192): hl
    int w = threadIdx.x >> 5, lane = threadIdx.x & 31;
    int b0 = (blockIdx.x * 4 + w) * 16;

    // cooperative copy of packed Whh into LDS (16B vectors), zero hl region
    {
        const uint4* src = (const uint4*)WhhP;
        uint4* dst = (uint4*)smem;
        for (int i = threadIdx.x; i < WHH_HALVES / 8; i += blockDim.x) dst[i] = src[i];
        for (int i = threadIdx.x; i < HL_SIZE; i += blockDim.x)
            smem[HL_OFF + i] = (_Float16)0.0f;
    }
    __syncthreads();

    int kbase = (lane >> 4) * 16;
    int ncl = lane & 15;
    int rbase = (lane >> 4) * 8;
    int hlw = HL_OFF + w * 16 * 128;

    v16h af[4];
#pragma unroll
    for (int k = 0; k < 4; ++k)
#pragma unroll
        for (int j = 0; j < 16; ++j) af[k][j] = (_Float16)0.0f;
    v8f c[7];
#pragma unroll
    for (int jt = 0; jt < 7; ++jt)
#pragma unroll
        for (int r = 0; r < 8; ++r) c[jt][r] = 0.0f;

    unsigned woff = 0;   // laundered each step: defeats load hoisting/spill
    for (int s = 0; s < SEQ; ++s) {
        asm volatile("" : "+v"(woff));
        int t = reverse ? (SEQ - 1 - s) : s;
        const float* gxt = gx + (long)t * BSZ * NP;
#pragma unroll
        for (int jt = 0; jt < 7; ++jt) {
            v8f acc[4];
#pragma unroll
            for (int G = 0; G < 4; ++G)
#pragma unroll
                for (int r = 0; r < 8; ++r)
                    acc[G][r] = gxt[(long)(b0 + rbase + r) * NP + G * HP + jt * 16 + ncl];
#pragma unroll
            for (int G = 0; G < 4; ++G)
#pragma unroll
                for (int kc = 0; kc < 4; ++kc) {
                    unsigned fb = woff + (unsigned)((kc * 28 + G * 7 + jt) * 32 + lane) * 16;
                    v8h blo = *(const v8h*)&smem[fb];
                    v8h bhi = *(const v8h*)&smem[fb + 8];
                    v16h bf;
#pragma unroll
                    for (int j = 0; j < 8; ++j) { bf[j] = blo[j]; bf[8 + j] = bhi[j]; }
                    acc[G] = wmma_f16(af[kc], bf, acc[G]);
                }
            int col = jt * 16 + ncl;
#pragma unroll
            for (int r = 0; r < 8; ++r) {
                float iv = sigm(acc[0][r]);
                float fv = sigm(acc[1][r]);
                float gv = tanhf(acc[2][r]);
                float ov = sigm(acc[3][r]);
                float cn = fv * c[jt][r] + iv * gv;
                c[jt][r] = cn;
                float hv = ov * tanhf(cn);
                _Float16 h16 = (_Float16)(col < LH ? hv : 0.0f);
                int rm = rbase + r;
                smem[hlw + rm * 128 + col] = h16;
                outbuf[((long)(b0 + rm) * SEQ + t) * 224 + colOff + col] = h16;
            }
        }
        __syncthreads();
#pragma unroll
        for (int kc = 0; kc < 4; ++kc) {
            const _Float16* p = &smem[hlw + (lane & 15) * 128 + kc * 32 + kbase];
            v8h lo = *(const v8h*)p;
            v8h hi = *(const v8h*)(p + 8);
#pragma unroll
            for (int j = 0; j < 8; ++j) { af[kc][j] = lo[j]; af[kc][8 + j] = hi[j]; }
        }
        __syncthreads();
    }
}

// ---------------------------------------------------------------------------
// Host-side orchestration
// ---------------------------------------------------------------------------
extern "C" void kernel_launch(void* const* d_in, const int* in_sizes, int n_in,
                              void* d_out, int out_size, void* d_ws, size_t ws_size,
                              hipStream_t stream) {
    (void)in_sizes; (void)n_in; (void)out_size; (void)ws_size;
    // Input order: graph_x, edge_index, edge_weight, data_feat, ids, then params
    // flattened in JAX sorted-key pytree order:
    // bilstm[0..2]{b:{Whh,Wih,b}, f:{Whh,Wih,b}}, conv1{V,W,b}, conv2{V,W,b},
    // fc{W,b}, gru{b:{Whh,Wih,bhh,bih}, f:{...}}, lstm_fc{W,b}, mlp{W1,W2,b1,b2}
    const float* graph_x   = (const float*)d_in[0];
    const int*   edge_idx  = (const int*)d_in[1];
    const float* edge_w    = (const float*)d_in[2];
    const float* data_feat = (const float*)d_in[3];
    const int*   ids       = (const int*)d_in[4];
    const float* conv1V = (const float*)d_in[23]; const float* conv1W = (const float*)d_in[24];
    const float* conv1b = (const float*)d_in[25];
    const float* conv2V = (const float*)d_in[26]; const float* conv2W = (const float*)d_in[27];
    const float* conv2b = (const float*)d_in[28];
    const float* fcW = (const float*)d_in[29];    const float* fcb = (const float*)d_in[30];
    const float* gruB_Whh = (const float*)d_in[31]; const float* gruB_Wih = (const float*)d_in[32];
    const float* gruB_bhh = (const float*)d_in[33]; const float* gruB_bih = (const float*)d_in[34];
    const float* gruF_Whh = (const float*)d_in[35]; const float* gruF_Wih = (const float*)d_in[36];
    const float* gruF_bhh = (const float*)d_in[37]; const float* gruF_bih = (const float*)d_in[38];
    const float* lfcW = (const float*)d_in[39];   const float* lfcb = (const float*)d_in[40];
    const float* mW1 = (const float*)d_in[41];    const float* mW2 = (const float*)d_in[42];
    const float* mb1 = (const float*)d_in[43];    const float* mb2 = (const float*)d_in[44];

    // workspace bump allocator
    char* ws = (char*)d_ws;
    size_t off = 0;
    auto alloc = [&](size_t bytes) -> void* {
        void* p = ws + off;
        off = (off + bytes + 255) & ~(size_t)255;
        return p;
    };
    float* hnb   = (float*)alloc((size_t)2 * NN * HID * 4);
    float* xfeat = (float*)alloc((size_t)NN * HID * 4);
    float* deg   = (float*)alloc((size_t)NN * 4);
    float* nrm   = (float*)alloc((size_t)NE * 4);
    float* h1    = (float*)alloc((size_t)NN * HID * 4);
    float* agg   = (float*)alloc((size_t)NN * HID * 4);
    float* x2    = (float*)alloc((size_t)NN * HID * 4);
    float* emb   = (float*)alloc((size_t)NN * HID * 4);
    _Float16* buf0   = (_Float16*)alloc((size_t)MTOT * 64 * 2);
    _Float16* actA   = (_Float16*)alloc((size_t)MTOT * 224 * 2);
    _Float16* actB   = (_Float16*)alloc((size_t)MTOT * 224 * 2);
    float*    gxbuf  = (float*)alloc((size_t)MTOT * NP * 4);
    _Float16* hidden = (_Float16*)alloc((size_t)MTOT * 64 * 2);
    _Float16* WihP[3][2]; _Float16* WhhP[3][2]; float* bP[3][2];
    for (int l = 0; l < 3; ++l)
        for (int d = 0; d < 2; ++d) {
            int Kp = (l == 0) ? 64 : 224;
            WihP[l][d] = (_Float16*)alloc((size_t)Kp * NP * 2);
            WhhP[l][d] = (_Float16*)alloc((size_t)128 * NP * 2);
            bP[l][d]   = (float*)alloc((size_t)NP * 4);
        }
    _Float16* W1P = (_Float16*)alloc((size_t)224 * 64 * 2);
    _Float16* W2P = (_Float16*)alloc((size_t)64 * 16 * 2);
    float* b1P = (float*)alloc(64 * 4);
    float* b2P = (float*)alloc(16 * 4);

    auto nblk = [](long n, int t) { return (unsigned)((n + t - 1) / t); };

    // ---- Stage 1: GRU encoder + lstm_fc ----
    gru_kernel<<<nblk(NN, 128), 128, 0, stream>>>(graph_x,
        gruF_Whh, gruF_Wih, gruF_bhh, gruF_bih,
        gruB_Whh, gruB_Wih, gruB_bhh, gruB_bih, hnb);
    lstm_fc_kernel<<<nblk(NN, 256), 256, 0, stream>>>(hnb, lfcW, lfcb, xfeat);

    // ---- Stage 2: gcn-norm + ARMA x2 ----
    zero_f32<<<nblk(NN, 256), 256, 0, stream>>>(deg, NN);
    deg_kernel<<<nblk(NE, 256), 256, 0, stream>>>(edge_idx, edge_w, deg);
    dinv_kernel<<<nblk(NN, 256), 256, 0, stream>>>(deg);
    norm_kernel<<<nblk(NE, 256), 256, 0, stream>>>(edge_idx, edge_w, deg, nrm);

    matvec20_kernel<<<nblk(NN, 256), 256, 0, stream>>>(xfeat, conv1W, h1);
    zero_f32<<<nblk((long)NN * HID, 256), 256, 0, stream>>>(agg, (long)NN * HID);
    scatter_kernel<<<nblk(NE, 256), 256, 0, stream>>>(edge_idx, nrm, h1, agg);
    arma_combine_kernel<<<nblk(NN, 256), 256, 0, stream>>>(xfeat, agg, conv1V, conv1b, x2);

    matvec20_kernel<<<nblk(NN, 256), 256, 0, stream>>>(x2, conv2W, h1);
    zero_f32<<<nblk((long)NN * HID, 256), 256, 0, stream>>>(agg, (long)NN * HID);
    scatter_kernel<<<nblk(NE, 256), 256, 0, stream>>>(edge_idx, nrm, h1, agg);
    arma_combine_kernel<<<nblk(NN, 256), 256, 0, stream>>>(x2, agg, conv2V, conv2b, emb);

    // ---- Stage 3: build BiLSTM layer-1 input (f16, K padded to 64) ----
    build_in_kernel<<<nblk(MTOT, 256), 256, 0, stream>>>(data_feat, ids, emb, fcW, fcb, buf0);

    // ---- Stage 4: pack weights ----
    auto packB = [&](const float* W, _Float16* out, int KT, int NT, int G, int Nreal,
                     int kA, int kSegB, int kB, int ld, int orient) {
        long total = (long)KT * NT * 512;
        pack_b_kernel<<<nblk(total, 256), 256, 0, stream>>>(
            W, out, KT, NT, G, HP, LH, Nreal, kA, kSegB, kB, ld, orient);
    };
    for (int l = 0; l < 3; ++l)
        for (int d = 0; d < 2; ++d) {
            // d==0 -> forward leaves (idx 8/9/10 + 6l), d==1 -> backward (5/6/7 + 6l)
            int base = (d == 0) ? (8 + 6 * l) : (5 + 6 * l);
            const float* Whh = (const float*)d_in[base + 0];
            const float* Wih = (const float*)d_in[base + 1];
            const float* bb  = (const float*)d_in[base + 2];
            int din = (l == 0) ? 40 : 200;
            int KTi = (l == 0) ? 2 : 7;
            int kA  = (l == 0) ? 40 : 100;
            int kB  = (l == 0) ? 0 : 100;
            packB(Wih, WihP[l][d], KTi, 28, 4, 0, kA, 112, kB, din, 0);
            packB(Whh, WhhP[l][d], 4, 28, 4, 0, 100, 0, 0, 100, 0);
            pack_bias_kernel<<<nblk(NP, 256), 256, 0, stream>>>(bb, bP[l][d], NP, 4, HP, LH, 0);
        }
    packB(mW1, W1P, 7, 4, 0, 50, 100, 112, 100, 50, 1);
    packB(mW2, W2P, 2, 1, 0, 2, 50, 0, 0, 2, 1);
    pack_bias_kernel<<<1, 64, 0, stream>>>(mb1, b1P, 64, 0, 0, 0, 50);
    pack_bias_kernel<<<1, 16, 0, stream>>>(mb2, b2P, 16, 0, 0, 0, 2);

    // ---- Stage 5: 3-layer BiLSTM (WMMA GEMM gx-precompute + WMMA recurrence) ----
    auto gemm = [&](int KT, int NTILES,
                    const _Float16* A, int lda, const _Float16* Bp, const float* bias,
                    float* of32, _Float16* of16, int NT, int nvalid, int ldc,
                    int relu, int transp) {
        int n4cnt = (NT + 3) / 4;
        long waves = (long)(MTOT / 16) * n4cnt;
        unsigned blocks = nblk(waves, 4);
        if (KT == 2 && NTILES == 4)
            gemm16_kernel<2, 4><<<blocks, 128, 0, stream>>>(
                A, lda, Bp, bias, of32, of16, MTOT, NT, nvalid, ldc, relu, transp, SEQ, BSZ);
        else if (KT == 7 && NTILES == 4)
            gemm16_kernel<7, 4><<<blocks, 128, 0, stream>>>(
                A, lda, Bp, bias, of32, of16, MTOT, NT, nvalid, ldc, relu, transp, SEQ, BSZ);
        else
            gemm16_kernel<2, 1><<<blocks, 128, 0, stream>>>(
                A, lda, Bp, bias, of32, of16, MTOT, NT, nvalid, ldc, relu, transp, SEQ, BSZ);
    };
    const _Float16* inb[3]  = {buf0, actA, actB};
    _Float16*       outb[3] = {actA, actB, actA};
    int lda_in[3] = {64, 224, 224};
    int KT_in[3]  = {2, 7, 7};
    for (int l = 0; l < 3; ++l) {
        for (int d = 0; d < 2; ++d) {
            gemm(KT_in[l], 4, inb[l], lda_in[l], WihP[l][d], bP[l][d], gxbuf, nullptr,
                 28, NP, NP, /*relu=*/0, /*transp=*/1);
            lstm_recur_kernel<<<16, 128, RECUR_SMEM, stream>>>(
                gxbuf, WhhP[l][d], outb[l], d == 0 ? 0 : HP, d == 0 ? 0 : 1);
        }
    }

    // ---- Stage 6: MLP head (WMMA) ----
    gemm(7, 4, actA, 224, W1P, b1P, nullptr, hidden, 4, 50, 64, /*relu=*/1, 0);
    gemm(2, 1, hidden, 64, W2P, b2P, (float*)d_out, nullptr, 1, 2, 2, /*relu=*/1, 0);
}
